// BertSelfAttention_22084721836055
// MI455X (gfx1250) — matrix operand order repaired and checked by reading, not compiled, not run on hardware
//
#include <hip/hip_runtime.h>

typedef __bf16 bf16_t;
typedef bf16_t v16bf __attribute__((ext_vector_type(16)));
typedef bf16_t v8bf  __attribute__((ext_vector_type(8)));
typedef float  v8f   __attribute__((ext_vector_type(8)));
typedef int    v4i   __attribute__((ext_vector_type(4)));

union V16U { v16bf v; v8bf h[2]; };

#if defined(__has_builtin)
#if __has_builtin(__builtin_amdgcn_global_load_async_to_lds_b128)
#define HAVE_ASYNC_LDS 1
#endif
#endif
#ifndef HAVE_ASYNC_LDS
#define HAVE_ASYNC_LDS 0
#endif

typedef v4i __attribute__((address_space(1)))* g4i_ptr;   // global int4*
typedef v4i __attribute__((address_space(3)))* l4i_ptr;   // LDS int4*

// 16-byte global -> LDS copy: async DMA (ASYNCcnt) when available, else direct.
__device__ __forceinline__ void copy16_to_lds(const bf16_t* g, bf16_t* l) {
#if HAVE_ASYNC_LDS
    __builtin_amdgcn_global_load_async_to_lds_b128(
        (g4i_ptr)(bf16_t*)g, (l4i_ptr)l, 0, 0);
#else
    *(v8bf*)l = *(const v8bf*)g;
#endif
}
__device__ __forceinline__ void wait_lds_copies() {
#if HAVE_ASYNC_LDS
    asm volatile("s_wait_asynccnt 0" ::: "memory");
#endif
}

__device__ __forceinline__ bf16_t f2bf(float f) {
    union { float f; unsigned u; } c; c.f = f;
    unsigned r = c.u + 0x7FFFu + ((c.u >> 16) & 1u);   // round-to-nearest-even
    union { unsigned short s; bf16_t b; } o; o.s = (unsigned short)(r >> 16);
    return o.b;
}

#define NBATCH 4
#define SEQ    1024
#define HID    1024
#define NHEAD  16
#define HDIM   64

// ---------------------------------------------------------------------------
// Kernel 0a: fp32 -> bf16 elementwise (X).  4 elements / thread.
// ---------------------------------------------------------------------------
__global__ __launch_bounds__(256)
void convert_bf16(const float* __restrict__ src, bf16_t* __restrict__ dst) {
    const size_t i = ((size_t)blockIdx.x * 256 + threadIdx.x) * 4;
    const float4 v = *(const float4*)(src + i);
    dst[i + 0] = f2bf(v.x);
    dst[i + 1] = f2bf(v.y);
    dst[i + 2] = f2bf(v.z);
    dst[i + 3] = f2bf(v.w);
}

// ---------------------------------------------------------------------------
// Kernel 0b: fp32 W[k][n] -> bf16 Wt[n][k] (32x32 LDS tile transpose).
// ---------------------------------------------------------------------------
__global__ __launch_bounds__(256)
void transpose_bf16(const float* __restrict__ W, bf16_t* __restrict__ Wt) {
    __shared__ float t[32][33];
    const int bx = blockIdx.x * 32;            // n base
    const int by = blockIdx.y * 32;            // k base
    const int tx = threadIdx.x & 31;
    const int ty = threadIdx.x >> 5;           // 0..7
    #pragma unroll
    for (int i = 0; i < 32; i += 8)
        t[ty + i][tx] = W[(size_t)(by + ty + i) * HID + bx + tx];
    __syncthreads();
    #pragma unroll
    for (int i = 0; i < 32; i += 8)
        Wt[(size_t)(bx + ty + i) * HID + by + tx] = f2bf(t[tx][ty + i]);
}

// ---------------------------------------------------------------------------
// Kernel 1: fused QKV projection. C = X @ Wt^T + b, bf16 out in [b,h,s,d].
// grid = (N/128, M/128, 3), block = 256 (8 waves). Double-buffered LDS,
// async global->LDS staging overlapped with WMMA compute.
// ---------------------------------------------------------------------------
__global__ __launch_bounds__(256)
void qkv_gemm(const bf16_t* __restrict__ X,
              const bf16_t* __restrict__ Wtq, const float* __restrict__ bq,
              const bf16_t* __restrict__ Wtk, const float* __restrict__ bk,
              const bf16_t* __restrict__ Wtv, const float* __restrict__ bv,
              bf16_t* __restrict__ Qb, bf16_t* __restrict__ Kb,
              bf16_t* __restrict__ Vb)
{
    const bf16_t* Wt; const float* bias; bf16_t* outp;
    if (blockIdx.z == 0)      { Wt = Wtq; bias = bq; outp = Qb; }
    else if (blockIdx.z == 1) { Wt = Wtk; bias = bk; outp = Kb; }
    else                      { Wt = Wtv; bias = bv; outp = Vb; }

    __shared__ bf16_t Atile[2][128][40];   // [m][k]
    __shared__ bf16_t Btile[2][128][40];   // [n][k] (W already transposed)

    const int tid  = threadIdx.x;
    const int lane = tid & 31;
    const int wave = tid >> 5;
    const int mBase = blockIdx.y * 128;
    const int nBase = blockIdx.x * 128;

    const int srow  = tid >> 1;            // 0..127
    const int scoff = (tid & 1) * 16;      // 0 / 16
    const bf16_t* gA = X  + (size_t)(mBase + srow) * HID + scoff;
    const bf16_t* gB = Wt + (size_t)(nBase + srow) * HID + scoff;

    v8f acc[8];
    #pragma unroll
    for (int n = 0; n < 8; ++n)
        #pragma unroll
        for (int e = 0; e < 8; ++e) acc[n][e] = 0.0f;

    // prologue: stage K-step 0 into buffer 0
    copy16_to_lds(gA,     &Atile[0][srow][scoff]);
    copy16_to_lds(gA + 8, &Atile[0][srow][scoff + 8]);
    copy16_to_lds(gB,     &Btile[0][srow][scoff]);
    copy16_to_lds(gB + 8, &Btile[0][srow][scoff + 8]);
    wait_lds_copies();
    __syncthreads();

    const int arow  = wave * 16 + (lane & 15);
    const int ac0   = (lane < 16) ? 0 : 8;
    const int bk0   = (lane < 16) ? 0 : 8;
    const int bcolL = lane & 15;

    #pragma unroll 1
    for (int idx = 0; idx < HID / 32; ++idx) {
        const int  cur     = idx & 1;
        const bool hasNext = (idx + 1) < (HID / 32);
        if (hasNext) {
            const int kk = (idx + 1) * 32;
            copy16_to_lds(gA + kk,     &Atile[cur ^ 1][srow][scoff]);
            copy16_to_lds(gA + kk + 8, &Atile[cur ^ 1][srow][scoff + 8]);
            copy16_to_lds(gB + kk,     &Btile[cur ^ 1][srow][scoff]);
            copy16_to_lds(gB + kk + 8, &Btile[cur ^ 1][srow][scoff + 8]);
        }

        V16U a;
        a.h[0] = *(const v8bf*)&Atile[cur][arow][ac0];
        a.h[1] = *(const v8bf*)&Atile[cur][arow][ac0 + 16];

        // two bursts of (gather 4 fragments, 4 WMMAs) to bound VGPR pressure
        #pragma unroll
        for (int g = 0; g < 2; ++g) {
            V16U bfr[4];
            #pragma unroll
            for (int n = 0; n < 4; ++n) {
                const int bn = g * 4 + n;
                bfr[n].h[0] = *(const v8bf*)&Btile[cur][bn * 16 + bcolL][bk0];
                bfr[n].h[1] = *(const v8bf*)&Btile[cur][bn * 16 + bcolL][bk0 + 16];
            }
            #pragma unroll
            for (int n = 0; n < 4; ++n)
                acc[g * 4 + n] = __builtin_amdgcn_wmma_f32_16x16x32_bf16(
                                     false, a.v, false, bfr[n].v, (short)0,
                                     acc[g * 4 + n], false, false);
        }

        if (hasNext) wait_lds_copies();
        __syncthreads();
    }

    // epilogue: + bias, bf16 store to [b, h, s, d]
    #pragma unroll
    for (int n = 0; n < 8; ++n) {
        const int j  = nBase + n * 16 + (lane & 15);
        const float bb = bias[j];
        const int h  = j >> 6, dp = j & 63;
        #pragma unroll
        for (int r = 0; r < 8; ++r) {
            const int m = mBase + wave * 16 + r + ((lane < 16) ? 0 : 8);
            const int bidx = m >> 10, s = m & 1023;
            const size_t oidx = ((size_t)(bidx * NHEAD + h) * SEQ + s) * HDIM + dp;
            outp[oidx] = f2bf(acc[n][r] + bb);
        }
    }
}

// ---------------------------------------------------------------------------
// Kernel 2: flash attention. grid = (S/128, B*NHEAD), block = 256 (8 waves).
// Double-buffered K/V staging overlapped with score/softmax/PV compute.
// ---------------------------------------------------------------------------
__global__ __launch_bounds__(256)
void attention(const bf16_t* __restrict__ Qb, const bf16_t* __restrict__ Kb,
               const bf16_t* __restrict__ Vb, float* __restrict__ Out)
{
    __shared__ bf16_t Kl[2][64][72];      // [key][d]
    __shared__ bf16_t Vt[2][64][72];      // [d][key]
    __shared__ bf16_t Pb[8][16][72];      // per-wave probs [row][key]

    const int tid  = threadIdx.x;
    const int lane = tid & 31;
    const int wave = tid >> 5;
    const int bh   = blockIdx.y;
    const int qBase = blockIdx.x * 128 + wave * 16;

    const bf16_t* Qh = Qb + (size_t)bh * SEQ * HDIM;
    const bf16_t* Kh = Kb + (size_t)bh * SEQ * HDIM;
    const bf16_t* Vh = Vb + (size_t)bh * SEQ * HDIM;

    // resident Q fragments
    V16U qa[2];
    {
        const int qrow = qBase + (lane & 15);
        const int c0   = (lane < 16) ? 0 : 8;
        const bf16_t* qp = Qh + (size_t)qrow * HDIM;
        qa[0].h[0] = *(const v8bf*)(qp + c0);
        qa[0].h[1] = *(const v8bf*)(qp + c0 + 16);
        qa[1].h[0] = *(const v8bf*)(qp + c0 + 32);
        qa[1].h[1] = *(const v8bf*)(qp + c0 + 48);
    }

    v8f o[4];
    float mrow[8], lrow[8];
    #pragma unroll
    for (int n = 0; n < 4; ++n)
        #pragma unroll
        for (int e = 0; e < 8; ++e) o[n][e] = 0.0f;
    #pragma unroll
    for (int r = 0; r < 8; ++r) { mrow[r] = -1e30f; lrow[r] = 0.0f; }

    const float kscale = 0.125f * 1.44269504088896340736f; // 1/sqrt(64)*log2(e)

    const int srow = tid >> 2;           // 0..63
    const int sc0  = (tid & 3) * 16;     // 0,16,32,48
    const int bk0  = (lane < 16) ? 0 : 8;
    const int colL = lane & 15;

    // prologue: stage key-block 0 into buffer 0
    {
        const bf16_t* ks = Kh + (size_t)srow * HDIM + sc0;
        copy16_to_lds(ks,     &Kl[0][srow][sc0]);
        copy16_to_lds(ks + 8, &Kl[0][srow][sc0 + 8]);
        const bf16_t* vs = Vh + (size_t)srow * HDIM + sc0;
        const v8bf v0 = *(const v8bf*)vs;
        const v8bf v1 = *(const v8bf*)(vs + 8);
        #pragma unroll
        for (int j = 0; j < 8; ++j) Vt[0][sc0 + j][srow]     = v0[j];
        #pragma unroll
        for (int j = 0; j < 8; ++j) Vt[0][sc0 + 8 + j][srow] = v1[j];
    }
    wait_lds_copies();
    __syncthreads();

    #pragma unroll 1
    for (int idx = 0; idx < SEQ / 64; ++idx) {
        const int  cur     = idx & 1;
        const bool hasNext = (idx + 1) < (SEQ / 64);
        if (hasNext) {
            const int kb = (idx + 1) * 64;
            const bf16_t* ks = Kh + (size_t)(kb + srow) * HDIM + sc0;
            copy16_to_lds(ks,     &Kl[cur ^ 1][srow][sc0]);
            copy16_to_lds(ks + 8, &Kl[cur ^ 1][srow][sc0 + 8]);
            const bf16_t* vs = Vh + (size_t)(kb + srow) * HDIM + sc0;
            const v8bf v0 = *(const v8bf*)vs;
            const v8bf v1 = *(const v8bf*)(vs + 8);
            #pragma unroll
            for (int j = 0; j < 8; ++j) Vt[cur ^ 1][sc0 + j][srow]     = v0[j];
            #pragma unroll
            for (int j = 0; j < 8; ++j) Vt[cur ^ 1][sc0 + 8 + j][srow] = v1[j];
        }

        // --- scores: two bursts of (4 fragments, 4 WMMAs) ---
        v8f st[4];
        #pragma unroll
        for (int n = 0; n < 4; ++n)
            #pragma unroll
            for (int e = 0; e < 8; ++e) st[n][e] = 0.0f;

        #pragma unroll
        for (int kh = 0; kh < 2; ++kh) {
            V16U kf[4];
            #pragma unroll
            for (int n = 0; n < 4; ++n) {
                const int d0 = kh * 32 + bk0;
                kf[n].h[0] = *(const v8bf*)&Kl[cur][n * 16 + colL][d0];
                kf[n].h[1] = *(const v8bf*)&Kl[cur][n * 16 + colL][d0 + 16];
            }
            #pragma unroll
            for (int n = 0; n < 4; ++n)
                st[n] = __builtin_amdgcn_wmma_f32_16x16x32_bf16(
                            false, qa[kh].v, false, kf[n].v, (short)0,
                            st[n], false, false);
        }

        // --- online softmax (row stats per 16-lane half) ---
        #pragma unroll
        for (int r = 0; r < 8; ++r) {
            float mt = -1e30f;
            #pragma unroll
            for (int n = 0; n < 4; ++n) mt = fmaxf(mt, st[n][r] * kscale);
            #pragma unroll
            for (int off = 8; off >= 1; off >>= 1)
                mt = fmaxf(mt, __shfl_xor(mt, off, 32));
            const float mnew  = fmaxf(mrow[r], mt);
            const float alpha = exp2f(mrow[r] - mnew);
            float rs = 0.0f;
            #pragma unroll
            for (int n = 0; n < 4; ++n) {
                const float p = exp2f(st[n][r] * kscale - mnew);
                st[n][r] = p;
                rs += p;
            }
            #pragma unroll
            for (int off = 8; off >= 1; off >>= 1)
                rs += __shfl_xor(rs, off, 32);
            lrow[r] = lrow[r] * alpha + rs;
            mrow[r] = mnew;
            #pragma unroll
            for (int n = 0; n < 4; ++n) o[n][r] *= alpha;
        }

        // --- probs to per-wave LDS region (C layout -> row major) ---
        {
            const int rbase = (lane < 16) ? 0 : 8;
            #pragma unroll
            for (int n = 0; n < 4; ++n)
                #pragma unroll
                for (int r = 0; r < 8; ++r)
                    Pb[wave][rbase + r][n * 16 + colL] = f2bf(st[n][r]);
        }
        asm volatile("s_wait_dscnt 0" ::: "memory");   // RAW on Pb within wave

        // --- P @ V: per kh, gather 4 V fragments then 4 WMMAs ---
        #pragma unroll
        for (int kh = 0; kh < 2; ++kh) {
            V16U ap;
            const int c0 = kh * 32 + ((lane < 16) ? 0 : 8);
            ap.h[0] = *(const v8bf*)&Pb[wave][colL][c0];
            ap.h[1] = *(const v8bf*)&Pb[wave][colL][c0 + 16];
            V16U vf[4];
            #pragma unroll
            for (int n = 0; n < 4; ++n) {
                const int k0 = kh * 32 + bk0;
                vf[n].h[0] = *(const v8bf*)&Vt[cur][n * 16 + colL][k0];
                vf[n].h[1] = *(const v8bf*)&Vt[cur][n * 16 + colL][k0 + 16];
            }
            #pragma unroll
            for (int n = 0; n < 4; ++n)
                o[n] = __builtin_amdgcn_wmma_f32_16x16x32_bf16(
                           false, ap.v, false, vf[n].v, (short)0, o[n], false, false);
        }

        if (hasNext) wait_lds_copies();
        __syncthreads();
    }

    // --- epilogue: normalize, fp32 store to [b, s, h*64+d] ---
    const int b = bh >> 4, h = bh & 15;
    #pragma unroll
    for (int n = 0; n < 4; ++n) {
        const int dcol = h * HDIM + n * 16 + colL;
        #pragma unroll
        for (int r = 0; r < 8; ++r) {
            const int s = qBase + r + ((lane < 16) ? 0 : 8);
            Out[((size_t)b * SEQ + s) * HID + dcol] = o[n][r] * (1.0f / lrow[r]);
        }
    }
}

// ---------------------------------------------------------------------------
extern "C" void kernel_launch(void* const* d_in, const int* in_sizes, int n_in,
                              void* d_out, int out_size, void* d_ws, size_t ws_size,
                              hipStream_t stream) {
    const float* X  = (const float*)d_in[0];
    const float* Wq = (const float*)d_in[1];
    const float* bq = (const float*)d_in[2];
    const float* Wk = (const float*)d_in[3];
    const float* bk = (const float*)d_in[4];
    const float* Wv = (const float*)d_in[5];
    const float* bv = (const float*)d_in[6];
    float* Out = (float*)d_out;

    const size_t qkvN = (size_t)NBATCH * NHEAD * SEQ * HDIM;   // 4M elements
    const size_t xN   = (size_t)NBATCH * SEQ * HID;            // 4M elements
    const size_t wN   = (size_t)HID * HID;                     // 1M elements

    bf16_t* Qb  = (bf16_t*)d_ws;
    bf16_t* Kb  = Qb  + qkvN;
    bf16_t* Vb  = Kb  + qkvN;
    bf16_t* Xbf = Vb  + qkvN;
    bf16_t* Wtq = Xbf + xN;
    bf16_t* Wtk = Wtq + wN;
    bf16_t* Wtv = Wtk + wN;

    convert_bf16<<<dim3((unsigned)(xN / 1024)), dim3(256), 0, stream>>>(X, Xbf);
    dim3 gT(HID / 32, HID / 32);
    transpose_bf16<<<gT, dim3(256), 0, stream>>>(Wq, Wtq);
    transpose_bf16<<<gT, dim3(256), 0, stream>>>(Wk, Wtk);
    transpose_bf16<<<gT, dim3(256), 0, stream>>>(Wv, Wtv);

    dim3 gGemm(HID / 128, (NBATCH * SEQ) / 128, 3);
    qkv_gemm<<<gGemm, dim3(256), 0, stream>>>(Xbf, Wtq, bq, Wtk, bk, Wtv, bv,
                                              Qb, Kb, Vb);

    dim3 gAttn(SEQ / 128, NBATCH * NHEAD);
    attention<<<gAttn, dim3(256), 0, stream>>>(Qb, Kb, Vb, Out);
}
